// RelationAttentionLayer_43293270343882
// MI455X (gfx1250) — matrix operand order repaired and checked by reading, not compiled, not run on hardware
//
#include <hip/hip_runtime.h>

// ---------------------------------------------------------------------------
// CDNA5 (gfx1250) fused relation-attention layer.
// D=768, H=12, HD=64, FFN=3072. Fuses token key/value projections (bf16 WMMA)
// with linear-normalized attention accumulation, output proj, LN, FFN, LN.
// Round 3: 64-token chunks, 4 M-tiles per wave sharing one B-fragment
// (quarters L2 weight traffic per WMMA vs round 1); epilogue LDS aliased into
// the tok staging buffer to fit 136 KB of the WGP's 320 KB LDS.
// ---------------------------------------------------------------------------

typedef __attribute__((ext_vector_type(16))) __bf16 v16bf;
typedef __attribute__((ext_vector_type(8)))  float  v8f;

union Frag16 {         // 16 bf16 = 32 bytes = two uint4
    uint4 u[2];
    v16bf v;
};

#define D_MODEL 768
#define N_HEAD  12
#define HD      64
#define FFN_DIM 3072
#define CHUNK   64     // tokens per WMMA chunk (four 16-row M-tiles)
#define MT      4      // M-tiles per wave

// ---------------------------------------------------------------------------
// Kernel 1: convert Wkt / Wvt (fp32 [in][out]) -> bf16 transposed [out][in]
// so WMMA B-fragments are contiguous 32-byte loads per lane (L2-resident).
// ---------------------------------------------------------------------------
__global__ void prep_weights(const float* __restrict__ Wkt,
                             const float* __restrict__ Wvt,
                             __bf16* __restrict__ WktT,
                             __bf16* __restrict__ WvtT) {
    int i = blockIdx.x * blockDim.x + threadIdx.x;   // over 768*768
    if (i < D_MODEL * D_MODEL) {
        int k = i / D_MODEL;       // input dim
        int n = i % D_MODEL;       // output dim
        WktT[(size_t)n * D_MODEL + k] = (__bf16)Wkt[i];
        WvtT[(size_t)n * D_MODEL + k] = (__bf16)Wvt[i];
    }
}

// ---------------------------------------------------------------------------
// Kernel 2: relation projections rq = rel@Wq+bq, rk = rel@Wkr+bkr,
// rv = rel@Wvr+bvr  (512x768 each; tiny -> fp32 VALU, rel row staged in LDS)
// ---------------------------------------------------------------------------
__global__ __launch_bounds__(256) void relproj(
    const float* __restrict__ rel,
    const float* __restrict__ Wq,  const float* __restrict__ bq,
    const float* __restrict__ Wvr, const float* __restrict__ bvr,
    const float* __restrict__ Wkr, const float* __restrict__ bkr,
    float* __restrict__ rq, float* __restrict__ rk, float* __restrict__ rv) {
    int b = blockIdx.x;
    int tid = threadIdx.x;
    __shared__ __align__(16) float r[D_MODEL];
    for (int i = tid; i < D_MODEL; i += 256) r[i] = rel[(size_t)b * D_MODEL + i];
    __syncthreads();
    for (int d = tid; d < D_MODEL; d += 256) {
        float aq = bq[d], ak = bkr[d], av = bvr[d];
        for (int e = 0; e < D_MODEL; ++e) {
            float x = r[e];
            aq += x * Wq [(size_t)e * D_MODEL + d];
            ak += x * Wkr[(size_t)e * D_MODEL + d];
            av += x * Wvr[(size_t)e * D_MODEL + d];
        }
        rq[(size_t)b * D_MODEL + d] = aq;
        rk[(size_t)b * D_MODEL + d] = ak;
        rv[(size_t)b * D_MODEL + d] = av;
    }
}

// ---------------------------------------------------------------------------
// Block-wide sum reduction (uniform control flow; red[] is 256 floats)
// ---------------------------------------------------------------------------
__device__ __forceinline__ float block_sum(float v, float* red, int tid) {
    red[tid] = v;
    __syncthreads();
    for (int s = 128; s > 0; s >>= 1) {
        if (tid < s) red[tid] += red[tid + s];
        __syncthreads();
    }
    float r = red[0];
    __syncthreads();
    return r;
}

// ---------------------------------------------------------------------------
// Kernel 3: fused attention + output proj + LN1 + FFN + LN2. One block per b.
// 256 threads = 8 waves. Waves 0-3 compute tk tiles, waves 4-7 tv tiles.
// Each wave: one 16-col N-tile of one matrix, four 16-row M-tiles (shared B).
// ---------------------------------------------------------------------------
__global__ __launch_bounds__(256) void fused_attn(
    const float* __restrict__ rel, const float* __restrict__ tok,
    const __bf16* __restrict__ WktT, const __bf16* __restrict__ WvtT,
    const float* __restrict__ bkt, const float* __restrict__ bvt,
    const float* __restrict__ rq,  const float* __restrict__ rk,
    const float* __restrict__ rv,
    const float* __restrict__ Wo,  const float* __restrict__ bo,
    const float* __restrict__ ln1g, const float* __restrict__ ln1b,
    const float* __restrict__ W1,  const float* __restrict__ b1,
    const float* __restrict__ W2,  const float* __restrict__ b2,
    const float* __restrict__ ln2g, const float* __restrict__ ln2b,
    float* __restrict__ out, int T) {

    const int b   = blockIdx.x;
    const int tid = threadIdx.x;
    const int wave = tid >> 5;
    const int lane = tid & 31;

    // 96 KB staging buffer; epilogue arrays alias it (tokA dead after chunks)
    __shared__ __align__(16) char bigbuf[CHUNK * D_MODEL * 2];
    __bf16* tokA = (__bf16*)bigbuf;                         // chunk stage (96 KB)
    float*  xrow = (float*)bigbuf;                          // epilogue: 768 f32
    float*  h1   = (float*)(bigbuf + 4096);                 // epilogue: 3072 f32
    float*  red  = (float*)(bigbuf + 4096 + FFN_DIM * 4);   // epilogue: 256 f32

    __shared__ __align__(16) float  tkv[2][CHUNK][66];      // tk / tv tiles (~33 KB)
    __shared__ __align__(16) float  accH[N_HEAD][HD];       // sum_t score*tv
    __shared__ float  ssum[N_HEAD];                         // sum_t score
    __shared__ float  sss[N_HEAD];                          // self score
    __shared__ __align__(16) float  rowA[D_MODEL];          // rq row -> update -> y

    // ---- init accumulators -------------------------------------------------
    for (int i = tid; i < N_HEAD * HD; i += 256) accH[i >> 6][i & 63] = 0.0f;
    if (tid < N_HEAD) ssum[tid] = 0.0f;

    // ---- async global->LDS staging of the rq row (ASYNCcnt path) -----------
    if (tid < D_MODEL / 4) {
        unsigned ldsoff = (unsigned)(size_t)(&rowA[tid * 4]);
        unsigned long long ga =
            (unsigned long long)(size_t)(rq + (size_t)b * D_MODEL + tid * 4);
        asm volatile("global_load_async_to_lds_b128 %0, %1, off"
                     :: "v"(ldsoff), "v"(ga) : "memory");
    }
    asm volatile("s_wait_asynccnt 0" ::: "memory");
    __syncthreads();

    // per-wave WMMA tile assignment (uniform within wave)
    const int mat = wave >> 2;            // 0 = key proj, 1 = value proj
    const int nt  = wave & 3;             // 16-col tile within the head's 64
    const __bf16* __restrict__ WT = mat ? WvtT : WktT;
    const int kbA = (lane >> 4) * 16;     // A-frag K sub-offset
    const int kbB = (lane >> 4) * 16;     // B-frag K sub-offset
    const int mrow = lane & 15;           // A-frag row within M-tile

    const int nChunks = T / CHUNK;
    for (int c = 0; c < nChunks; ++c) {
        // ---- stage tok chunk [64 x 768] fp32 -> bf16 LDS -------------------
        const float4* tg =
            (const float4*)(tok + ((size_t)b * T + (size_t)c * CHUNK) * D_MODEL);
        for (int i = tid; i < CHUNK * D_MODEL / 4; i += 256) {
            float4 v = tg[i];
            int base = i * 4;
            tokA[base + 0] = (__bf16)v.x;
            tokA[base + 1] = (__bf16)v.y;
            tokA[base + 2] = (__bf16)v.z;
            tokA[base + 3] = (__bf16)v.w;
        }
        __syncthreads();

        for (int h = 0; h < N_HEAD; ++h) {
            // ---- 64x64 GEMM tile for this head: 4 WMMAs share one B-frag ---
            const int ncol = h * HD + nt * 16 + (lane & 15);
            const __bf16* __restrict__ wrow = WT + (size_t)ncol * D_MODEL;
            v8f acc[MT];
#pragma unroll
            for (int mt = 0; mt < MT; ++mt) acc[mt] = (v8f){};
            for (int k0 = 0; k0 < D_MODEL; k0 += 32) {
                Frag16 a[MT], w;
                const uint4* br = (const uint4*)(wrow + k0 + kbB);
                w.u[0] = br[0];
                w.u[1] = br[1];
#pragma unroll
                for (int mt = 0; mt < MT; ++mt) {
                    const uint4* ar = (const uint4*)
                        (tokA + (mt * 16 + mrow) * D_MODEL + k0 + kbA);
                    a[mt].u[0] = ar[0];   // K = k0+kb .. +7
                    a[mt].u[1] = ar[1];   // K = k0+kb+8 .. +15
                }
                // prefetch this weight row 256 elements (8 K-tiles) ahead into cache
                __builtin_prefetch((const void*)(wrow + k0 + 256), 0, 1);
#pragma unroll
                for (int mt = 0; mt < MT; ++mt)
                    acc[mt] = __builtin_amdgcn_wmma_f32_16x16x32_bf16(
                        false, a[mt].v, false, w.v, (short)0, acc[mt], false, false);
            }
            // ---- scatter accumulators to LDS (m = r + 8*(lane>>4), n = lane&15)
            {
                const int row0 = (lane >> 4) * 8;
                const int col  = nt * 16 + (lane & 15);
#pragma unroll
                for (int mt = 0; mt < MT; ++mt)
#pragma unroll
                    for (int r2 = 0; r2 < 8; ++r2)
                        tkv[mat][mt * 16 + row0 + r2][col] = acc[mt][r2];
            }
            __syncthreads();

            // ---- scores + weighted value accumulation (4 thr per token) ----
            {
                const int t = tid >> 2;     // token within chunk (0..63)
                const int g = tid & 3;      // 16-feature group
                float p = 0.0f;
#pragma unroll
                for (int j = 0; j < 16; ++j) {
                    int d2 = g * 16 + j;
                    p += rowA[h * HD + d2] * (tkv[0][t][d2] + bkt[h * HD + d2]);
                }
                // butterfly across the 4-lane group -> everyone has score
                for (int off = 2; off >= 1; off >>= 1)
                    p += __shfl_xor(p, off, 4);
                float score = p;
                if (g == 0) atomicAdd(&ssum[h], score);
#pragma unroll
                for (int j = 0; j < 16; ++j) {
                    int d2 = g * 16 + j;
                    atomicAdd(&accH[h][d2],
                              score * (tkv[1][t][d2] + bvt[h * HD + d2]));
                }
            }
            __syncthreads();
        }
        __syncthreads();
    }

    // ---- self score per head ----------------------------------------------
    if (tid < N_HEAD) {
        float s = 0.0f;
        for (int d2 = 0; d2 < HD; ++d2)
            s += rowA[tid * HD + d2] * rk[(size_t)b * D_MODEL + tid * HD + d2];
        sss[tid] = s;
    }
    __syncthreads();

    // ---- update = (self*rv + acc) / denom  (reuse rowA) --------------------
    for (int d = tid; d < D_MODEL; d += 256) {
        int h2 = d >> 6;
        float denom = sss[h2] + ssum[h2];
        float u = (sss[h2] * rv[(size_t)b * D_MODEL + d] + accH[h2][d & 63]) / denom;
        rowA[d] = u;
    }
    __syncthreads();

    // ---- output projection + residual (xrow aliases dead tokA buffer) ------
    for (int d = tid; d < D_MODEL; d += 256) {
        float a = bo[d];
        for (int e = 0; e < D_MODEL; ++e)
            a += rowA[e] * Wo[(size_t)e * D_MODEL + d];
        xrow[d] = rel[(size_t)b * D_MODEL + d] + a;
    }
    __syncthreads();

    // ---- LayerNorm 1 --------------------------------------------------------
    {
        float ls = 0.0f;
        for (int d = tid; d < D_MODEL; d += 256) ls += xrow[d];
        float mu = block_sum(ls, red, tid) * (1.0f / D_MODEL);
        float lv = 0.0f;
        for (int d = tid; d < D_MODEL; d += 256) {
            float t2 = xrow[d] - mu;
            lv += t2 * t2;
        }
        float var = block_sum(lv, red, tid) * (1.0f / D_MODEL);
        float rstd = rsqrtf(var + 1e-5f);
        for (int d = tid; d < D_MODEL; d += 256)
            xrow[d] = (xrow[d] - mu) * rstd * ln1g[d] + ln1b[d];
    }
    __syncthreads();

    // ---- FFN ----------------------------------------------------------------
    for (int f = tid; f < FFN_DIM; f += 256) {
        float a = b1[f];
        for (int e = 0; e < D_MODEL; ++e)
            a += xrow[e] * W1[(size_t)e * FFN_DIM + f];
        h1[f] = fmaxf(a, 0.0f);
    }
    __syncthreads();
    for (int d = tid; d < D_MODEL; d += 256) {
        float a = b2[d];
        for (int f = 0; f < FFN_DIM; ++f)
            a += h1[f] * W2[(size_t)f * D_MODEL + d];
        rowA[d] = xrow[d] + a;            // y row (reuse rowA)
    }
    __syncthreads();

    // ---- LayerNorm 2 + store -----------------------------------------------
    {
        float ls = 0.0f;
        for (int d = tid; d < D_MODEL; d += 256) ls += rowA[d];
        float mu = block_sum(ls, red, tid) * (1.0f / D_MODEL);
        float lv = 0.0f;
        for (int d = tid; d < D_MODEL; d += 256) {
            float t2 = rowA[d] - mu;
            lv += t2 * t2;
        }
        float var = block_sum(lv, red, tid) * (1.0f / D_MODEL);
        float rstd = rsqrtf(var + 1e-5f);
        for (int d = tid; d < D_MODEL; d += 256)
            out[(size_t)b * D_MODEL + d] =
                (rowA[d] - mu) * rstd * ln2g[d] + ln2b[d];
    }
}

// ---------------------------------------------------------------------------
// Host launcher
// ---------------------------------------------------------------------------
extern "C" void kernel_launch(void* const* d_in, const int* in_sizes, int n_in,
                              void* d_out, int out_size, void* d_ws, size_t ws_size,
                              hipStream_t stream) {
    const float* rel  = (const float*)d_in[0];
    const float* tok  = (const float*)d_in[1];
    const float* Wq   = (const float*)d_in[2];
    const float* bq   = (const float*)d_in[3];
    const float* Wvr  = (const float*)d_in[4];
    const float* bvr  = (const float*)d_in[5];
    const float* Wkr  = (const float*)d_in[6];
    const float* bkr  = (const float*)d_in[7];
    const float* Wkt  = (const float*)d_in[8];
    const float* bkt  = (const float*)d_in[9];
    const float* Wvt  = (const float*)d_in[10];
    const float* bvt  = (const float*)d_in[11];
    const float* Wo   = (const float*)d_in[12];
    const float* bo   = (const float*)d_in[13];
    const float* ln1g = (const float*)d_in[14];
    const float* ln1b = (const float*)d_in[15];
    const float* W1   = (const float*)d_in[16];
    const float* b1   = (const float*)d_in[17];
    const float* W2   = (const float*)d_in[18];
    const float* b2   = (const float*)d_in[19];
    const float* ln2g = (const float*)d_in[20];
    const float* ln2b = (const float*)d_in[21];

    const int B = in_sizes[0] / D_MODEL;
    const long long tokN = (long long)in_sizes[1];
    const int T = (int)(tokN / ((long long)B * D_MODEL));

    // workspace layout
    char* ws = (char*)d_ws;
    const size_t wbytes = (size_t)D_MODEL * D_MODEL * sizeof(__bf16);
    __bf16* WktT = (__bf16*)ws;            ws += wbytes;
    __bf16* WvtT = (__bf16*)ws;            ws += wbytes;
    const size_t rbytes = (size_t)B * D_MODEL * sizeof(float);
    float* rq = (float*)ws;                ws += rbytes;
    float* rk = (float*)ws;                ws += rbytes;
    float* rv = (float*)ws;                ws += rbytes;

    prep_weights<<<(D_MODEL * D_MODEL + 255) / 256, 256, 0, stream>>>(
        Wkt, Wvt, WktT, WvtT);

    relproj<<<B, 256, 0, stream>>>(rel, Wq, bq, Wvr, bvr, Wkr, bkr, rq, rk, rv);

    fused_attn<<<B, 256, 0, stream>>>(
        rel, tok, WktT, WvtT, bkt, bvt, rq, rk, rv,
        Wo, bo, ln1g, ln1b, W1, b1, W2, b2, ln2g, ln2b,
        (float*)d_out, T);
}
